// MultiHeadSelfAttention_5446018531426
// MI455X (gfx1250) — matrix operand order repaired and checked by reading, compiled, not run on hardware
//
#include <hip/hip_runtime.h>

// ---------------- CDNA5 WMMA plumbing ----------------
typedef __bf16 bf16;
typedef __attribute__((ext_vector_type(16))) __bf16 v16bf;
typedef __attribute__((ext_vector_type(8)))  float  v8f;
typedef int b128_t __attribute__((vector_size(16)));  // matches builtin's V4i

union FragU { v16bf v; uint4 q[2]; };

#define SEQ 512
#define DM  512
#define HD  64
#define NH  8
#define XQ_LD  520  // 16 rows * 520 * 2B = 16640 B per wave (16B-aligned, bank-staggered)
#define KEY_LD 520  // 32 rows * 520 * 2B = 33280 B per buffer
#define P_LD   40   // 16 rows * 40  * 2B = 1280 B per wave

#if __has_builtin(__builtin_amdgcn_global_load_async_to_lds_b128)
#define HAS_ASYNC 1
#else
#define HAS_ASYNC 0
#endif

#if __has_builtin(__builtin_amdgcn_s_wait_asynccnt)
#define WAIT_ASYNC(n) __builtin_amdgcn_s_wait_asynccnt(n)
#elif HAS_ASYNC
#define WAIT_ASYNC(n) asm volatile("s_wait_asynccnt %0" ::"i"(n) : "memory")
#else
#define WAIT_ASYNC(n)
#endif

// Pointer laundering: makes loads derived from p non-loop-invariant so LICM
// cannot hoist whole fragment sets out of outer loops (spill prevention).
template <typename T>
__device__ __forceinline__ const T* launder(const T* p) {
  asm volatile("" : "+v"(p));
  return p;
}

__device__ __forceinline__ v8f wmma_bf16(v16bf a, v16bf b, v8f c) {
  return __builtin_amdgcn_wmma_f32_16x16x32_bf16(false, a, false, b, (short)0, c,
                                                 false, false);
}

// A fragment: 16x32 tile from row-major [M x K] source (stride ld elements).
template <typename PT>
__device__ __forceinline__ v16bf load_frag_a(PT base, int row0, int k0, int ld, int lane) {
  const int m = lane & 15, half = lane >> 4;
  PT p = base + (row0 + m) * ld + k0 + half * 8;
  FragU f;
  f.q[0] = *(const uint4*)(p);
  f.q[1] = *(const uint4*)(p + 16);
  return f.v;
}

// B fragment: 32x16 tile B[k,n] from transposed (row-major N x K) storage.
template <typename PT>
__device__ __forceinline__ v16bf load_frag_b(PT base, int n0, int k0, int ld, int lane) {
  const int n = lane & 15, half = lane >> 4;
  PT p = base + (n0 + n) * ld + k0 + half * 8;
  FragU f;
  f.q[0] = *(const uint4*)(p);
  f.q[1] = *(const uint4*)(p + 16);
  return f.v;
}

__device__ __forceinline__ float redmax16(float v) {
  v = fmaxf(v, __shfl_xor(v, 1, 32));
  v = fmaxf(v, __shfl_xor(v, 2, 32));
  v = fmaxf(v, __shfl_xor(v, 4, 32));
  v = fmaxf(v, __shfl_xor(v, 8, 32));
  return v;
}
__device__ __forceinline__ float redsum16(float v) {
  v += __shfl_xor(v, 1, 32);
  v += __shfl_xor(v, 2, 32);
  v += __shfl_xor(v, 4, 32);
  v += __shfl_xor(v, 8, 32);
  return v;
}

// Cooperative async copy of 32 key rows (32x512 bf16 = 32 KB) into padded LDS.
// 8 async b128 instructions per wave -> tracked on ASYNCcnt.
__device__ __forceinline__ void stage_keys(const bf16* __restrict__ xB, int j0,
                                           bf16* buf, int tid) {
#pragma unroll
  for (int i = 0; i < 8; ++i) {
    const int off = (i * 256 + tid) * 8;       // element offset in 32x512 tile
    const int row = off >> 9, col = off & 511;
    const bf16* g = xB + (size_t)(j0 + row) * DM + col;
    bf16* l = buf + row * KEY_LD + col;
#if HAS_ASYNC
    __builtin_amdgcn_global_load_async_to_lds_b128(
        (__attribute__((address_space(1))) b128_t*)(b128_t*)g,
        (__attribute__((address_space(3))) b128_t*)(b128_t*)l, 0, 0);
#else
    *(uint4*)l = *(const uint4*)g;
#endif
  }
}

// ---------------- staging kernels ----------------
__global__ void cvt_x_kernel(const float* __restrict__ x, bf16* __restrict__ xb, int n) {
  int i = (blockIdx.x * blockDim.x + threadIdx.x) * 4;
  if (i < n) {
    float4 v = *(const float4*)(x + i);
    xb[i + 0] = (bf16)v.x; xb[i + 1] = (bf16)v.y;
    xb[i + 2] = (bf16)v.z; xb[i + 3] = (bf16)v.w;
  }
}

// Qt[h][d][e] = Q[h][e][d]  (bf16)
__global__ void tq_kernel(const float* __restrict__ q, bf16* __restrict__ qt) {
  int idx = blockIdx.x * blockDim.x + threadIdx.x;
  int e = idx & 511;
  int d = (idx >> 9) & 511;
  int h = idx >> 18;
  qt[idx] = (bf16)q[(h << 18) + (e << 9) + d];
}

// Vt[h][c][e] = V[h][e][c]  (bf16)
__global__ void tv_kernel(const float* __restrict__ v, bf16* __restrict__ vt) {
  int idx = blockIdx.x * blockDim.x + threadIdx.x;
  int e = idx & 511;
  int c = (idx >> 9) & 63;
  int h = idx >> 15;
  vt[idx] = (bf16)v[(h << 15) + (e << 6) + c];
}

// xvT[b][h][col][n] = (x[b] @ V_h)[n][col] via WMMA.
// One 16-row strip per wave (no outer loop -> nothing loop-invariant to hoist).
// grid = B*H*(SEQ/128), 256 threads = 8 waves.
__global__ __launch_bounds__(256, 1)
void mhsa_xv_kernel(const bf16* __restrict__ xb, const bf16* __restrict__ vt,
                    bf16* __restrict__ xvt) {
  const int lane = threadIdx.x & 31;
  const int wave = threadIdx.x >> 5;
  const int blk = blockIdx.x;
  const int rowblk = blk & 3;
  const int h = (blk >> 2) & 7;
  const int b = blk >> 5;
  const bf16* xB  = xb + (size_t)b * SEQ * DM;
  const bf16* vtH = vt + (size_t)h * HD * DM;
  bf16* outBH = xvt + ((size_t)(b * NH + h)) * HD * SEQ;
  const int g = lane >> 4, nl = lane & 15;
  const int row0 = rowblk * 128 + wave * 16;

  v8f acc0 = {}, acc1 = {}, acc2 = {}, acc3 = {};
  v16bf a  = load_frag_a(xB, row0, 0, DM, lane);
  v16bf b0 = load_frag_b(vtH,  0, 0, DM, lane);
  v16bf b1 = load_frag_b(vtH, 16, 0, DM, lane);
  v16bf b2 = load_frag_b(vtH, 32, 0, DM, lane);
  v16bf b3 = load_frag_b(vtH, 48, 0, DM, lane);
  for (int kt = 0; kt < 16; ++kt) {
    v16bf an = a, bn0 = b0, bn1 = b1, bn2 = b2, bn3 = b3;
    if (kt + 1 < 16) {                    // issue next-k loads before this k's WMMAs
      const int k1 = (kt + 1) * 32;
      an  = load_frag_a(xB, row0, k1, DM, lane);
      bn0 = load_frag_b(vtH,  0, k1, DM, lane);
      bn1 = load_frag_b(vtH, 16, k1, DM, lane);
      bn2 = load_frag_b(vtH, 32, k1, DM, lane);
      bn3 = load_frag_b(vtH, 48, k1, DM, lane);
    }
    acc0 = wmma_bf16(a, b0, acc0);
    acc1 = wmma_bf16(a, b1, acc1);
    acc2 = wmma_bf16(a, b2, acc2);
    acc3 = wmma_bf16(a, b3, acc3);
    a = an; b0 = bn0; b1 = bn1; b2 = bn2; b3 = bn3;
  }
#pragma unroll
  for (int r = 0; r < 8; ++r) {
    const int row = row0 + g * 8 + r;
    outBH[(size_t)( 0 + nl) * SEQ + row] = (bf16)acc0[r];
    outBH[(size_t)(16 + nl) * SEQ + row] = (bf16)acc1[r];
    outBH[(size_t)(32 + nl) * SEQ + row] = (bf16)acc2[r];
    outBH[(size_t)(48 + nl) * SEQ + row] = (bf16)acc3[r];
  }
}

// ---------------- main flash-attention kernel ----------------
// grid = B*H*(SEQ/128); 256 threads = 8 waves; wave owns 16 query rows.
__global__ __launch_bounds__(256, 1)
void mhsa_attn_kernel(const bf16* __restrict__ xb, const bf16* __restrict__ qt,
                      const bf16* __restrict__ xvt, float* __restrict__ out) {
  __shared__ bf16 xq_lds[8][16 * XQ_LD];   // 133120 B
  __shared__ bf16 key_lds[2][32 * KEY_LD]; //  66560 B (async double buffer)
  __shared__ bf16 p_lds[8][16 * P_LD];     //  10240 B

  const int tid  = threadIdx.x;
  const int lane = tid & 31;
  const int wave = tid >> 5;
  const int blk = blockIdx.x;
  const int rowblk = blk & 3;
  const int h = (blk >> 2) & 7;
  const int b = blk >> 5;
  const int row0 = rowblk * 128 + wave * 16;

  const bf16* xB    = xb  + (size_t)b * SEQ * DM;
  const bf16* qtH   = qt  + (size_t)h * DM * DM;
  const bf16* xvtBH = xvt + ((size_t)(b * NH + h)) * HD * SEQ;
  bf16* myxq = &xq_lds[wave][0];
  bf16* myp  = &p_lds[wave][0];
  const int g = lane >> 4, nl = lane & 15;

  // Kick off async copy of the first key tile; it completes under phase A.
  stage_keys(xB, 0, &key_lds[0][0], tid);

  // Phase A: XQ strip (16 x 512) = x[rows] @ Q_h -> LDS (bf16).
  // 4 accumulators per pass; next-k fragments issued before current WMMAs.
  for (int nt4 = 0; nt4 < 8; ++nt4) {
    const int n0 = nt4 * 64;
    const bf16* xBl = launder(xB);   // block A-fragment hoisting across nt4
    v8f acc0 = {}, acc1 = {}, acc2 = {}, acc3 = {};
    v16bf a  = load_frag_a(xBl, row0, 0, DM, lane);
    v16bf b0 = load_frag_b(qtH, n0 +  0, 0, DM, lane);
    v16bf b1 = load_frag_b(qtH, n0 + 16, 0, DM, lane);
    v16bf b2 = load_frag_b(qtH, n0 + 32, 0, DM, lane);
    v16bf b3 = load_frag_b(qtH, n0 + 48, 0, DM, lane);
    for (int kt = 0; kt < 16; ++kt) {
      v16bf an = a, bn0 = b0, bn1 = b1, bn2 = b2, bn3 = b3;
      if (kt + 1 < 16) {
        const int k1 = (kt + 1) * 32;
        an  = load_frag_a(xBl, row0, k1, DM, lane);
        bn0 = load_frag_b(qtH, n0 +  0, k1, DM, lane);
        bn1 = load_frag_b(qtH, n0 + 16, k1, DM, lane);
        bn2 = load_frag_b(qtH, n0 + 32, k1, DM, lane);
        bn3 = load_frag_b(qtH, n0 + 48, k1, DM, lane);
      }
      acc0 = wmma_bf16(a, b0, acc0);
      acc1 = wmma_bf16(a, b1, acc1);
      acc2 = wmma_bf16(a, b2, acc2);
      acc3 = wmma_bf16(a, b3, acc3);
      a = an; b0 = bn0; b1 = bn1; b2 = bn2; b3 = bn3;
    }
#pragma unroll
    for (int r = 0; r < 8; ++r) {
      const int rr = (g * 8 + r) * XQ_LD + n0 + nl;
      myxq[rr +  0] = (bf16)acc0[r];
      myxq[rr + 16] = (bf16)acc1[r];
      myxq[rr + 32] = (bf16)acc2[r];
      myxq[rr + 48] = (bf16)acc3[r];
    }
  }
  __syncthreads();

  // Phase B: flash attention over key chunks of 32, keys staged in LDS.
  v8f o0 = {}, o1 = {}, o2 = {}, o3 = {};
  float mrun[8], lrun[8];
#pragma unroll
  for (int r = 0; r < 8; ++r) { mrun[r] = -1e30f; lrun[r] = 0.0f; }

  for (int jc = 0; jc < 16; ++jc) {
    const int j0 = jc * 32;
    if (jc + 1 < 16) {
      stage_keys(xB, j0 + 32, &key_lds[(jc + 1) & 1][0], tid);  // overlap next tile
      WAIT_ASYNC(8);   // first-issued 8 (current tile) complete; 8 still in flight
    } else {
      WAIT_ASYNC(0);
    }
    __syncthreads();
    const bf16* kb = &key_lds[jc & 1][0];
    const bf16* xqL = launder((const bf16*)myxq);  // block A-frag hoisting across jc

    v8f s0 = {}, s1 = {};
#pragma unroll 4
    for (int kt = 0; kt < 16; ++kt) {
      v16bf a  = load_frag_a(xqL, 0, kt * 32, XQ_LD, lane);    // ds
      v16bf b0 = load_frag_b(kb,  0, kt * 32, KEY_LD, lane);   // ds
      v16bf b1 = load_frag_b(kb, 16, kt * 32, KEY_LD, lane);   // ds
      s0 = wmma_bf16(a, b0, s0);
      s1 = wmma_bf16(a, b1, s1);
    }

    // Pre-load xv B-fragments; their global latency hides under softmax VALU.
    v16bf vb0 = load_frag_b(xvtBH,  0, j0, SEQ, lane);
    v16bf vb1 = load_frag_b(xvtBH, 16, j0, SEQ, lane);
    v16bf vb2 = load_frag_b(xvtBH, 32, j0, SEQ, lane);
    v16bf vb3 = load_frag_b(xvtBH, 48, j0, SEQ, lane);

    // Online softmax (rows g*8+r, replicated across the 16 lanes of each group).
#pragma unroll
    for (int r = 0; r < 8; ++r) {
      float tmax = redmax16(fmaxf(s0[r], s1[r]));
      float mnew = fmaxf(mrun[r], tmax);
      float scale = __expf(mrun[r] - mnew);
      float p0 = __expf(s0[r] - mnew);
      float p1 = __expf(s1[r] - mnew);
      float rsum = redsum16(p0 + p1);
      lrun[r] = lrun[r] * scale + rsum;
      mrun[r] = mnew;
      o0[r] *= scale; o1[r] *= scale; o2[r] *= scale; o3[r] *= scale;
      myp[(g * 8 + r) * P_LD + nl]      = (bf16)p0;  // D-layout -> LDS
      myp[(g * 8 + r) * P_LD + 16 + nl] = (bf16)p1;
    }

    // O += P @ xv  (per-wave LDS; DS ops are in-order within a wave)
    v16bf pa = load_frag_a((const bf16*)myp, 0, 0, P_LD, lane);
    o0 = wmma_bf16(pa, vb0, o0);
    o1 = wmma_bf16(pa, vb1, o1);
    o2 = wmma_bf16(pa, vb2, o2);
    o3 = wmma_bf16(pa, vb3, o3);

    __syncthreads();  // all waves done with kb before it is overwritten
  }

  // normalize and write out[b, row, h*64 + col]  (f32)
  float* outB = out + ((size_t)b * SEQ + row0) * DM + h * HD;
#pragma unroll
  for (int r = 0; r < 8; ++r) {
    const float inv = 1.0f / lrun[r];
    float* po = outB + (size_t)(g * 8 + r) * DM;
    po[ 0 + nl] = o0[r] * inv;
    po[16 + nl] = o1[r] * inv;
    po[32 + nl] = o2[r] * inv;
    po[48 + nl] = o3[r] * inv;
  }
}

// ---------------- launcher ----------------
extern "C" void kernel_launch(void* const* d_in, const int* in_sizes, int n_in,
                              void* d_out, int out_size, void* d_ws, size_t ws_size,
                              hipStream_t stream) {
  const float* x = (const float*)d_in[0];
  const float* Q = (const float*)d_in[1];
  const float* V = (const float*)d_in[2];
  float* out = (float*)d_out;

  char* ws = (char*)d_ws;
  size_t off = 0;
  bf16* xb  = (bf16*)(ws + off); off += (size_t)32 * SEQ * DM * 2;        // 16 MB
  bf16* qt  = (bf16*)(ws + off); off += (size_t)NH * DM * DM * 2;         //  4 MB
  bf16* vt  = (bf16*)(ws + off); off += (size_t)NH * HD * DM * 2;         // 0.5 MB
  bf16* xvt = (bf16*)(ws + off); off += (size_t)32 * NH * HD * SEQ * 2;   // 16 MB

  const int n_x = 32 * SEQ * DM;  // 8388608
  cvt_x_kernel<<<n_x / (256 * 4), 256, 0, stream>>>(x, xb, n_x);
  tq_kernel<<<(NH * DM * DM) / 256, 256, 0, stream>>>(Q, qt);
  tv_kernel<<<(NH * HD * DM) / 256, 256, 0, stream>>>(V, vt);
  mhsa_xv_kernel<<<32 * NH * (SEQ / 128), 256, 0, stream>>>(xb, vt, xvt);
  mhsa_attn_kernel<<<32 * NH * (SEQ / 128), 256, 0, stream>>>(xb, qt, xvt, out);
}
